// HANLayer_46952582480385
// MI455X (gfx1250) — matrix operand order repaired and checked by reading, not compiled, not run on hardware
//
#include <hip/hip_runtime.h>
#include <hip/hip_bf16.h>

#define BS 8
#define NR 2000
#define NC 2000
#define EMB 128
#define KC 32
#define NCHUNK ((NC + KC - 1) / KC)   // 63
#define VTS 40                        // ldsVT stride in halves (32 + 8 pad)

#if defined(__has_builtin)
#  if __has_builtin(__builtin_amdgcn_global_load_async_to_lds_b128)
#    define HAVE_ASYNC 1
#  endif
#  if __has_builtin(__builtin_amdgcn_s_wait_asynccnt)
#    define HAVE_WAIT_ASYNC 1
#  endif
#endif
#ifndef HAVE_ASYNC
#  define HAVE_ASYNC 0
#endif
#ifndef HAVE_WAIT_ASYNC
#  define HAVE_WAIT_ASYNC 0
#endif

typedef __attribute__((ext_vector_type(16))) _Float16 v16h;
typedef __attribute__((ext_vector_type(8)))  _Float16 v8h;
typedef __attribute__((ext_vector_type(2)))  _Float16 h2;
typedef __attribute__((ext_vector_type(8)))  float    v8f;
typedef __attribute__((ext_vector_type(4)))  float    f4;

// async-to-LDS builtin parameter types: V4i addrspace(1)* , V4i addrspace(3)*
typedef int v4i __attribute__((vector_size(16)));
typedef __attribute__((address_space(1))) v4i* as1_v4i;
typedef __attribute__((address_space(3))) v4i* as3_v4i;

__device__ __forceinline__ float lrelu(float x) { return fmaxf(x, 0.01f * x); }

__device__ __forceinline__ void wait_async0() {
#if HAVE_ASYNC
#  if HAVE_WAIT_ASYNC
    __builtin_amdgcn_s_wait_asynccnt(0);
#  else
    asm volatile("s_wait_asynccnt 0" ::: "memory");
#  endif
#endif
}

// Fill one wave's 16x32 f32 cost tile in LDS for chunk starting at kc0.
// Async path: 4 x global_load_async_to_lds_b128 per lane (no VGPR round-trip),
// tracked by ASYNCcnt. OOB columns are clamped (values masked to -inf later).
__device__ __forceinline__ void cost_fill(float (*__restrict__ ldsCw)[KC + 4],
                                          const float* __restrict__ costB,
                                          int kc0, int lane)
{
#if HAVE_ASYNC
    #pragma unroll
    for (int i = 0; i < 4; ++i) {
        const int s   = i * 32 + lane;
        const int row = s >> 3;
        const int sg  = s & 7;
        int c = kc0 + sg * 4;
        c = c > NC - 4 ? NC - 4 : c;            // clamp, garbage masked later
        const float* g  = costB + (size_t)row * NC + c;
        float*       lp = &ldsCw[row][sg * 4];
        __builtin_amdgcn_global_load_async_to_lds_b128(
            (as1_v4i)(void*)g, (as3_v4i)(void*)lp, 0, 0);
    }
#else
    #pragma unroll
    for (int it = 0; it < 4; ++it) {
        const int rr = it * 4 + (lane >> 3);
        const int cq = (lane & 7) * 4;
        f4 z = {};
        f4 v = (kc0 + cq < NC) ? *(const f4*)(costB + (size_t)rr * NC + kc0 + cq) : z;
        *(f4*)&ldsCw[rr][cq] = v;
    }
#endif
}

// ---------------------------------------------------------------------------
// Projection kernel: one wave per dot product of length EMB=128.
// ---------------------------------------------------------------------------
__global__ __launch_bounds__(256)
void proj_kernel(const float* __restrict__ row_emb,
                 const float* __restrict__ col_emb,
                 const float* __restrict__ W,
                 float* __restrict__ rp,    // [BS*NR]
                 float* __restrict__ sa,    // [BS*NR]  leaky(self_logit)
                 float* __restrict__ cpj)   // [BS*NC]
{
    const int lane = threadIdx.x & 31;
    const int wid  = (blockIdx.x * blockDim.x + threadIdx.x) >> 5;

    f4 wr = *(const f4*)(W + lane * 4);
    f4 wc = *(const f4*)(W + EMB + lane * 4);

    if (wid < BS * NR) {
        f4 ev = *(const f4*)(row_emb + (size_t)wid * EMB + lane * 4);
        float d1 = ev[0]*wr[0] + ev[1]*wr[1] + ev[2]*wr[2] + ev[3]*wr[3];
        float d2 = ev[0]*wc[0] + ev[1]*wc[1] + ev[2]*wc[2] + ev[3]*wc[3];
        #pragma unroll
        for (int off = 16; off > 0; off >>= 1) {
            d1 += __shfl_xor(d1, off, 32);
            d2 += __shfl_xor(d2, off, 32);
        }
        if (lane == 0) {
            rp[wid] = d1;
            sa[wid] = lrelu(d1 + d2);
        }
    } else if (wid < BS * (NR + NC)) {
        const int idx = wid - BS * NR;
        f4 ev = *(const f4*)(col_emb + (size_t)idx * EMB + lane * 4);
        float d2 = ev[0]*wc[0] + ev[1]*wc[1] + ev[2]*wc[2] + ev[3]*wc[3];
        #pragma unroll
        for (int off = 16; off > 0; off >>= 1) d2 += __shfl_xor(d2, off, 32);
        if (lane == 0) cpj[idx] = d2;
    }
}

// ---------------------------------------------------------------------------
// Fused online-softmax attention: single pass over cost_mat, WMMA P@V,
// async double-buffered cost tiles.
// ---------------------------------------------------------------------------
__global__ __launch_bounds__(128)
void han_attn_kernel(const float* __restrict__ row_emb,
                     const float* __restrict__ col_emb,
                     const float* __restrict__ cost,
                     const float* __restrict__ rp,
                     const float* __restrict__ sa,
                     const float* __restrict__ cpj,
                     float* __restrict__ out)
{
    __shared__ _Float16 ldsVT[EMB * VTS];         // [emb][col] f16 chunk
    __shared__ float    ldsC[2][4][16][KC + 4];   // double-buffered cost tiles
    __shared__ float    ldsS[4][16];              // per-row broadcast
    __shared__ float    ldsS2[4][16];             // per-row broadcast (1/l)

    const int tid  = threadIdx.x;
    const int lane = tid & 31;
    const int w    = tid >> 5;
    const int h    = lane >> 4;        // half-wave select
    const int r    = lane & 15;
    const int b    = blockIdx.y;

    int tile = blockIdx.x * 4 + w;
    tile = tile > (NR / 16 - 1) ? (NR / 16 - 1) : tile;   // clamp (dup ok)
    const int rowbase = tile * 16;
    const int grow    = rowbase + r;

    const float rpv = rp[b * NR + grow];
    const float sav = sa[b * NR + grow];
    float m = sav;          // softmax running max, seeded with self logit
    float l = 1.0f;         // running sum (self term = exp(0) = 1)

    v8f acc[8];
    #pragma unroll
    for (int t = 0; t < 8; ++t) acc[t] = (v8f){};

    const float* costB = cost + ((size_t)b * NR + rowbase) * NC;
    const float* cpB   = cpj + b * NC;
    const float* vB    = col_emb + (size_t)b * NC * EMB;

    const int cpair = tid & 15;   // column pair for VT fill
    const int eg    = tid >> 4;   // emb group (16 embs)

    int buf = 0;
    cost_fill(ldsC[0][w], costB, 0, lane);   // prologue: chunk 0 in flight

    for (int ch = 0; ch < NCHUNK; ++ch) {
        const int kc0 = ch * KC;
        __syncthreads();   // previous chunk's VT reads complete

        // --- cooperative fill of ldsVT: col_emb[kc0..kc0+31][:] -> f16 [emb][col]
        {
            const int c0 = kc0 + 2 * cpair;
            const bool v0 = (c0 < NC), v1 = (c0 + 1 < NC);
            const float* p0 = vB + (size_t)c0 * EMB + eg * 16;
            const float* p1 = p0 + EMB;
            #pragma unroll
            for (int q = 0; q < 4; ++q) {
                f4 z = {};
                f4 a0 = v0 ? *(const f4*)(p0 + q * 4) : z;
                f4 a1 = v1 ? *(const f4*)(p1 + q * 4) : z;
                const int e = eg * 16 + q * 4;
                #pragma unroll
                for (int j = 0; j < 4; ++j) {
                    h2 pk = { (_Float16)a0[j], (_Float16)a1[j] };
                    *(h2*)(ldsVT + (size_t)(e + j) * VTS + 2 * cpair) = pk;
                }
            }
        }
        wait_async0();     // this wave's cost tile (chunk ch) has landed in LDS
        __syncthreads();   // VT ready

        // --- software pipeline: launch next chunk's cost tile now
        if (ch + 1 < NCHUNK) cost_fill(ldsC[buf ^ 1][w], costB, kc0 + KC, lane);

        // --- per-lane logits -> online softmax probabilities (A operand layout)
        const int cb1 = kc0 + 8 * h;          // K = 8h + j     (j = 0..7)
        const int cb2 = kc0 + 16 + 8 * h;     // K = 16 + 8h + j
        const bool ok1 = (cb1 < NC), ok2 = (cb2 < NC);
        f4 z4 = {};
        f4 c1a = *(const f4*)&ldsC[buf][w][r][8 * h];
        f4 c1b = *(const f4*)&ldsC[buf][w][r][8 * h + 4];
        f4 c2a = *(const f4*)&ldsC[buf][w][r][16 + 8 * h];
        f4 c2b = *(const f4*)&ldsC[buf][w][r][16 + 8 * h + 4];
        f4 q1a = ok1 ? *(const f4*)(cpB + cb1)     : z4;
        f4 q1b = ok1 ? *(const f4*)(cpB + cb1 + 4) : z4;
        f4 q2a = ok2 ? *(const f4*)(cpB + cb2)     : z4;
        f4 q2b = ok2 ? *(const f4*)(cpB + cb2 + 4) : z4;

        float act1[8], act2[8];
        const float NEG = -__builtin_inff();
        #pragma unroll
        for (int j = 0; j < 4; ++j) {
            act1[j]     = ok1 ? lrelu((rpv + q1a[j]) * c1a[j]) : NEG;
            act1[4 + j] = ok1 ? lrelu((rpv + q1b[j]) * c1b[j]) : NEG;
            act2[j]     = ok2 ? lrelu((rpv + q2a[j]) * c2a[j]) : NEG;
            act2[4 + j] = ok2 ? lrelu((rpv + q2b[j]) * c2b[j]) : NEG;
        }
        float lmax = NEG;
        #pragma unroll
        for (int j = 0; j < 8; ++j) lmax = fmaxf(lmax, fmaxf(act1[j], act2[j]));
        lmax = fmaxf(lmax, __shfl_xor(lmax, 16, 32));   // combine lane pair (row)
        const float m_new = fmaxf(m, lmax);
        const float s = __expf(m - m_new);
        m = m_new;

        float sum = 0.0f;
        v16h A;
        #pragma unroll
        for (int j = 0; j < 8; ++j) {
            float e1 = __expf(act1[j] - m_new);
            float e2 = __expf(act2[j] - m_new);
            sum += e1 + e2;
            A[j]     = (_Float16)e1;
            A[8 + j] = (_Float16)e2;
        }
        sum += __shfl_xor(sum, 16, 32);
        l = l * s + sum;

        // --- broadcast per-row rescale factor, rescale accumulators
        if (lane < 16) ldsS[w][r] = s;
        __syncthreads();
        f4 f0 = *(const f4*)&ldsS[w][8 * h];
        f4 f1 = *(const f4*)&ldsS[w][8 * h + 4];
        float fr[8] = { f0[0], f0[1], f0[2], f0[3], f1[0], f1[1], f1[2], f1[3] };
        #pragma unroll
        for (int t = 0; t < 8; ++t)
            #pragma unroll
            for (int i = 0; i < 8; ++i)
                acc[t][i] *= fr[i];

        // --- 8 WMMA tiles: D(16x16 f32) += A(16x32 f16) x B(32x16 f16)
        #pragma unroll
        for (int t = 0; t < 8; ++t) {
            const _Float16* bp = ldsVT + (size_t)(16 * t + r) * VTS + 8 * h;
            v8h blo = *(const v8h*)bp;        // K = 8h .. 8h+7   of column N=r
            v8h bhi = *(const v8h*)(bp + 16); // K = 16+8h .. 16+8h+7
            v16h B;
            #pragma unroll
            for (int j = 0; j < 8; ++j) { B[j] = blo[j]; B[8 + j] = bhi[j]; }
            acc[t] = __builtin_amdgcn_wmma_f32_16x16x32_f16(
                false, A, false, B, (short)0, acc[t], false, false);
        }
        buf ^= 1;
    }

    // --- epilogue: out = (acc + p_self * row_emb) / l
    const float pself = __expf(sav - m);
    const float linv  = 1.0f / l;
    __syncthreads();
    if (lane < 16) { ldsS[w][r] = pself; ldsS2[w][r] = linv; }
    __syncthreads();
    f4 p0 = *(const f4*)&ldsS[w][8 * h];
    f4 p1 = *(const f4*)&ldsS[w][8 * h + 4];
    f4 l0 = *(const f4*)&ldsS2[w][8 * h];
    f4 l1 = *(const f4*)&ldsS2[w][8 * h + 4];
    float psv[8] = { p0[0], p0[1], p0[2], p0[3], p1[0], p1[1], p1[2], p1[3] };
    float liv[8] = { l0[0], l0[1], l0[2], l0[3], l1[0], l1[1], l1[2], l1[3] };

    const float* reB  = row_emb + ((size_t)b * NR + rowbase) * EMB;
    float*       outB = out     + ((size_t)b * NR + rowbase) * EMB;
    #pragma unroll
    for (int i = 0; i < 8; ++i) {
        const int row = 8 * h + i;
        #pragma unroll
        for (int t = 0; t < 8; ++t) {
            const int n = 16 * t + r;
            const float re = reB[(size_t)row * EMB + n];
            outB[(size_t)row * EMB + n] = (acc[t][i] + psv[i] * re) * liv[i];
        }
    }
}

// ---------------------------------------------------------------------------
extern "C" void kernel_launch(void* const* d_in, const int* in_sizes, int n_in,
                              void* d_out, int out_size, void* d_ws, size_t ws_size,
                              hipStream_t stream) {
    const float* row_emb = (const float*)d_in[0];   // [BS,NR,EMB]
    const float* col_emb = (const float*)d_in[1];   // [BS,NC,EMB]
    const float* cost    = (const float*)d_in[2];   // [BS,NR,NC]
    const float* W       = (const float*)d_in[3];   // [2*EMB]
    float* out = (float*)d_out;

    float* rp  = (float*)d_ws;            // [BS*NR]
    float* sa  = rp + BS * NR;            // [BS*NR]
    float* cpj = sa + BS * NR;            // [BS*NC]

    const int dots   = BS * (NR + NC);    // 32000 wave-dot-products
    const int wpb    = 256 / 32;
    const int blocks = (dots + wpb - 1) / wpb;
    proj_kernel<<<blocks, 256, 0, stream>>>(row_emb, col_emb, W, rp, sa, cpj);

    dim3 grid((NR / 16 + 3) / 4, BS);     // 32 x 8, 4 wave-tiles per block
    han_attn_kernel<<<grid, 128, 0, stream>>>(row_emb, col_emb, cost,
                                              rp, sa, cpj, out);
}
